// SeqToSeq_29506425324288
// MI455X (gfx1250) — matrix operand order repaired and checked by reading, not compiled, not run on hardware
//
#include <hip/hip_runtime.h>

// ---------- constants ----------
#define BSZ   8192
#define TT    7
#define UU    356
#define NG    1424      // 4*U
#define FE    69
#define FD    45
#define KE    448       // pad(69+356=425 -> mult 32)
#define KD    416       // pad(45+356=401 -> mult 32)
#define NPADL 1536      // pad 1424 -> mult 128
#define KMAP  2496      // pad 7*356=2492 -> mult 32
#define NH    1024
#define NOUT  168
#define NOUTP 256       // pad 168 -> mult 128

typedef __attribute__((ext_vector_type(16))) __bf16 v16bf;
typedef __attribute__((ext_vector_type(8)))  float  v8f;

__device__ __forceinline__ unsigned short f2bf(float f) {
  unsigned u = __float_as_uint(f);
  unsigned r = u + 0x7FFFu + ((u >> 16) & 1u);
  return (unsigned short)(r >> 16);
}
__device__ __forceinline__ float sigf(float x) { return 1.0f / (1.0f + __expf(-x)); }

union FragAB { v16bf bf; uint4 u4[2]; };

// ---------- generic bf16 WMMA GEMM:  out = act(A[MxK] * BT[NxK]^T + bias) ----------
// A row-major bf16 (lda = Kpad), BT row-major bf16 (ldb = Kpad, rows = padded N, zero filled)
__global__ void __launch_bounds__(128)
gemm_bf16_wmma(const unsigned short* __restrict__ A, int lda,
               const unsigned short* __restrict__ BT, int ldb,
               const float* __restrict__ bias,
               float* __restrict__ outF, unsigned short* __restrict__ outBF,
               int N, int nK, int act) {
  const int lane = threadIdx.x & 31;
  const int wave = threadIdx.x >> 5;
  const int wm = wave & 1, wn = wave >> 1;
  const int nl = lane & 15, hi = lane >> 4;
  const int blockM = blockIdx.x * 64;
  const int blockN = blockIdx.y * 128;

  v8f zero = {};
  v8f acc[2][4];
#pragma unroll
  for (int i = 0; i < 2; ++i)
#pragma unroll
    for (int j = 0; j < 4; ++j) acc[i][j] = zero;

  const unsigned short* aRow[2];
#pragma unroll
  for (int mt = 0; mt < 2; ++mt)
    aRow[mt] = A + (size_t)(blockM + wm * 32 + mt * 16 + nl) * lda;
  const unsigned short* bRow[4];
#pragma unroll
  for (int nt = 0; nt < 4; ++nt)
    bRow[nt] = BT + (size_t)(blockN + wn * 64 + nt * 16 + nl) * ldb;

  for (int kk = 0; kk < nK; ++kk) {
    const int k0 = kk * 32;
    FragAB a[2], b[4];
#pragma unroll
    for (int mt = 0; mt < 2; ++mt) {
      a[mt].u4[0] = *(const uint4*)(aRow[mt] + k0 + hi * 8);        // K: hi*8..+7
      a[mt].u4[1] = *(const uint4*)(aRow[mt] + k0 + 16 + hi * 8);   // K: 16+hi*8..+7
    }
#pragma unroll
    for (int nt = 0; nt < 4; ++nt) {
      b[nt].u4[0] = *(const uint4*)(bRow[nt] + k0 + hi * 8);        // K: hi*8..+7
      b[nt].u4[1] = *(const uint4*)(bRow[nt] + k0 + 16 + hi * 8);   // K: 16+hi*8..+7
    }
    if (kk + 1 < nK) {  // CDNA5 global_prefetch_b8 for next A K-slab
      __builtin_prefetch(aRow[0] + k0 + 32, 0, 0);
      __builtin_prefetch(aRow[1] + k0 + 32, 0, 0);
    }
#pragma unroll
    for (int mt = 0; mt < 2; ++mt)
#pragma unroll
      for (int nt = 0; nt < 4; ++nt)
        acc[mt][nt] = __builtin_amdgcn_wmma_f32_16x16x32_bf16(
            false, a[mt].bf, false, b[nt].bf, (short)0, acc[mt][nt], false, false);
  }

#pragma unroll
  for (int mt = 0; mt < 2; ++mt)
#pragma unroll
    for (int nt = 0; nt < 4; ++nt) {
      const int cn = blockN + wn * 64 + nt * 16 + nl;
      if (cn >= N) continue;
      const float bv = bias ? bias[cn] : 0.0f;
#pragma unroll
      for (int v = 0; v < 8; ++v) {
        const int r = blockM + wm * 32 + mt * 16 + hi * 8 + v;
        float val = acc[mt][nt][v] + bv;
        if (act == 1) val = val > 0.0f ? val : 0.0f;
        else if (act == 2) val = tanhf(val);
        if (outF)  outF[(size_t)r * N + cn] = val;
        if (outBF) outBF[(size_t)r * N + cn] = f2bf(val);
      }
    }
}

// ---------- LSTM gate fusion ----------
__global__ void __launch_bounds__(256)
lstm_gate(const float* __restrict__ z, float* __restrict__ c,
          unsigned short* __restrict__ hdst, int h_ld, int h_col,
          unsigned short* __restrict__ dec, int t) {
  int idx = blockIdx.x * blockDim.x + threadIdx.x;
  if (idx >= BSZ * UU) return;
  int b = idx / UU, u = idx - b * UU;
  const float* zb = z + (size_t)b * NG;
  float zi = zb[u], zf = zb[UU + u], zg = zb[2 * UU + u], zo = zb[3 * UU + u];
  float cn = sigf(zf) * c[idx] + sigf(zi) * tanhf(zg);
  float hn = sigf(zo) * tanhf(cn);
  c[idx] = cn;
  unsigned short hb = f2bf(hn);
  hdst[(size_t)b * h_ld + h_col + u] = hb;
  if (dec) dec[(size_t)b * KMAP + t * UU + u] = hb;
}

// ---------- pack x_t into concat-A buffer (cols [0,F) data, [F+U,Kpad) zero) ----------
__global__ void __launch_bounds__(256)
pack_x(const float* __restrict__ x, int F, int t,
       unsigned short* __restrict__ Adst, int Kpad, int FU, int total) {
  int idx = blockIdx.x * blockDim.x + threadIdx.x;
  if (idx >= total) return;
  int j = idx % Kpad, b = idx / Kpad;
  if (j < F)       Adst[idx] = f2bf(x[((size_t)b * TT + t) * F + j]);
  else if (j >= FU) Adst[idx] = 0;
}

// ---------- weight converters (transpose f32 [KxN] -> bf16 [Npad x Kpad]) ----------
__global__ void __launch_bounds__(256)
conv_wt(const float* __restrict__ W, int K, int N,
        unsigned short* __restrict__ dst, int Kpad, int total) {
  int idx = blockIdx.x * blockDim.x + threadIdx.x;
  if (idx >= total) return;
  int k = idx % Kpad, n = idx / Kpad;
  float v = (k < K && n < N) ? W[(size_t)k * N + n] : 0.0f;
  dst[idx] = f2bf(v);
}
__global__ void __launch_bounds__(256)
conv_wcat(const float* __restrict__ W, const float* __restrict__ Wr, int F, int N,
          unsigned short* __restrict__ dst, int Kpad, int total) {
  int idx = blockIdx.x * blockDim.x + threadIdx.x;
  if (idx >= total) return;
  int k = idx % Kpad, n = idx / Kpad;
  float v = 0.0f;
  if (n < N) {
    if (k < F)            v = W[(size_t)k * N + n];
    else if (k < F + UU)  v = Wr[(size_t)(k - F) * N + n];
  }
  dst[idx] = f2bf(v);
}

// ---------- state init: c=0, h-region of A_enc=0, decoded pad cols=0 ----------
__global__ void __launch_bounds__(256)
init_state(float* __restrict__ c, unsigned short* __restrict__ Aenc) {
  int idx = blockIdx.x * blockDim.x + threadIdx.x;
  if (idx >= BSZ * UU) return;
  int b = idx / UU, u = idx - b * UU;
  c[idx] = 0.0f;
  Aenc[(size_t)b * KE + FE + u] = 0;
}
__global__ void __launch_bounds__(256)
init_decpad(unsigned short* __restrict__ dec) {
  int idx = blockIdx.x * blockDim.x + threadIdx.x;
  if (idx >= BSZ * 4) return;
  int b = idx / 4, j = idx - b * 4;
  dec[(size_t)b * KMAP + (TT * UU) + j] = 0;
}

extern "C" void kernel_launch(void* const* d_in, const int* in_sizes, int n_in,
                              void* d_out, int out_size, void* d_ws, size_t ws_size,
                              hipStream_t stream) {
  const float* x     = (const float*)d_in[0];
  const float* m     = (const float*)d_in[1];
  const float* enc_W = (const float*)d_in[2];
  const float* enc_U = (const float*)d_in[3];
  const float* enc_b = (const float*)d_in[4];
  const float* dec_W = (const float*)d_in[5];
  const float* dec_U = (const float*)d_in[6];
  const float* dec_b = (const float*)d_in[7];
  const float* W_map = (const float*)d_in[8];
  const float* b_map = (const float*)d_in[9];
  const float* W1    = (const float*)d_in[10];
  const float* b1    = (const float*)d_in[11];
  const float* W2    = (const float*)d_in[12];
  const float* b2    = (const float*)d_in[13];
  const float* W3    = (const float*)d_in[14];
  const float* b3    = (const float*)d_in[15];
  const float* W_out = (const float*)d_in[16];
  const float* b_out = (const float*)d_in[17];
  float* out = (float*)d_out;

  // workspace carve-up (deterministic, 256B aligned)
  size_t off = 0;
  auto carve = [&](size_t bytes) { size_t o = off; off = (off + bytes + 255) & ~(size_t)255; return o; };
  char* ws = (char*)d_ws;
  unsigned short* WcatE = (unsigned short*)(ws + carve((size_t)NPADL * KE * 2));
  unsigned short* WcatD = (unsigned short*)(ws + carve((size_t)NPADL * KD * 2));
  unsigned short* WmapT = (unsigned short*)(ws + carve((size_t)NH * KMAP * 2));
  unsigned short* W1T   = (unsigned short*)(ws + carve((size_t)NH * NH * 2));
  unsigned short* W2T   = (unsigned short*)(ws + carve((size_t)NH * NH * 2));
  unsigned short* W3T   = (unsigned short*)(ws + carve((size_t)NH * NH * 2));
  unsigned short* WoutT = (unsigned short*)(ws + carve((size_t)NOUTP * NH * 2));
  unsigned short* Aenc  = (unsigned short*)(ws + carve((size_t)BSZ * KE * 2));
  unsigned short* Adec  = (unsigned short*)(ws + carve((size_t)BSZ * KD * 2));
  float*          Z     = (float*)         (ws + carve((size_t)BSZ * NG * 4));
  float*          Cbuf  = (float*)         (ws + carve((size_t)BSZ * UU * 4));
  unsigned short* Dec   = (unsigned short*)(ws + carve((size_t)BSZ * KMAP * 2));
  unsigned short* Act0  = (unsigned short*)(ws + carve((size_t)BSZ * NH * 2));
  unsigned short* Act1  = (unsigned short*)(ws + carve((size_t)BSZ * NH * 2));
  (void)ws_size; (void)n_in; (void)in_sizes; (void)out_size;

  const dim3 blk256(256);
  auto g1 = [](int total) { return dim3((total + 255) / 256); };

  // weight conversion / transpose to bf16
  conv_wcat<<<g1(NPADL * KE), blk256, 0, stream>>>(enc_W, enc_U, FE, NG, WcatE, KE, NPADL * KE);
  conv_wcat<<<g1(NPADL * KD), blk256, 0, stream>>>(dec_W, dec_U, FD, NG, WcatD, KD, NPADL * KD);
  conv_wt<<<g1(NH * KMAP), blk256, 0, stream>>>(W_map, TT * UU, NH, WmapT, KMAP, NH * KMAP);
  conv_wt<<<g1(NH * NH), blk256, 0, stream>>>(W1, NH, NH, W1T, NH, NH * NH);
  conv_wt<<<g1(NH * NH), blk256, 0, stream>>>(W2, NH, NH, W2T, NH, NH * NH);
  conv_wt<<<g1(NH * NH), blk256, 0, stream>>>(W3, NH, NH, W3T, NH, NH * NH);
  conv_wt<<<g1(NOUTP * NH), blk256, 0, stream>>>(W_out, NH, NOUT, WoutT, NH, NOUTP * NH);

  init_state<<<g1(BSZ * UU), blk256, 0, stream>>>(Cbuf, Aenc);
  init_decpad<<<g1(BSZ * 4), blk256, 0, stream>>>(Dec);

  const dim3 gblk(128);
  const dim3 gridL(BSZ / 64, (NG + 127) / 128);      // 128 x 12
  const dim3 gridH(BSZ / 64, (NH + 127) / 128);      // 128 x 8
  const dim3 gridO(BSZ / 64, (NOUT + 127) / 128);    // 128 x 2

  // encoder: z = [x_t, h] @ WcatE + b ; gates
  for (int t = 0; t < TT; ++t) {
    pack_x<<<g1(BSZ * KE), blk256, 0, stream>>>(x, FE, t, Aenc, KE, FE + UU, BSZ * KE);
    gemm_bf16_wmma<<<gridL, gblk, 0, stream>>>(Aenc, KE, WcatE, KE, enc_b, Z, nullptr, NG, KE / 32, 0);
    unsigned short* hdst = (t < TT - 1) ? Aenc : Adec;     // final h seeds decoder A
    lstm_gate<<<g1(BSZ * UU), blk256, 0, stream>>>(Z, Cbuf, hdst,
        (t < TT - 1) ? KE : KD, (t < TT - 1) ? FE : FD, nullptr, 0);
  }
  // decoder: z = [m_t, h] @ WcatD + b ; gates ; h -> decoded
  for (int t = 0; t < TT; ++t) {
    pack_x<<<g1(BSZ * KD), blk256, 0, stream>>>(m, FD, t, Adec, KD, FD + UU, BSZ * KD);
    gemm_bf16_wmma<<<gridL, gblk, 0, stream>>>(Adec, KD, WcatD, KD, dec_b, Z, nullptr, NG, KD / 32, 0);
    lstm_gate<<<g1(BSZ * UU), blk256, 0, stream>>>(Z, Cbuf, Adec, KD, FD, Dec, t);
  }
  // MLP head
  gemm_bf16_wmma<<<gridH, gblk, 0, stream>>>(Dec,  KMAP, WmapT, KMAP, b_map, nullptr, Act0, NH, KMAP / 32, 1);
  gemm_bf16_wmma<<<gridH, gblk, 0, stream>>>(Act0, NH,   W1T,   NH,   b1,    nullptr, Act1, NH, NH / 32,   2);
  gemm_bf16_wmma<<<gridH, gblk, 0, stream>>>(Act1, NH,   W2T,   NH,   b2,    nullptr, Act0, NH, NH / 32,   2);
  gemm_bf16_wmma<<<gridH, gblk, 0, stream>>>(Act0, NH,   W3T,   NH,   b3,    nullptr, Act1, NH, NH / 32,   2);
  gemm_bf16_wmma<<<gridO, gblk, 0, stream>>>(Act1, NH,   WoutT, NH,   b_out, out,     nullptr, NOUT, NH / 32, 0);
}
